// ContactGraspModel_5257039970861
// MI455X (gfx1250) — compile-verified, matrix operand order checked by reading, not run on hardware
//
#include <hip/hip_runtime.h>
#include <hip/hip_bf16.h>

typedef __attribute__((ext_vector_type(16))) _Float16 v16h;
typedef __attribute__((ext_vector_type(8)))  float    v8f;
typedef __attribute__((ext_vector_type(4)))  unsigned int v4u;
typedef __attribute__((ext_vector_type(8)))  int      v8i;
typedef __attribute__((ext_vector_type(4)))  int      v4i;

#define BATCH 4
#define NPTS0 3000
#define NP1   1000
#define NP2   512

#if __has_builtin(__builtin_amdgcn_tensor_load_to_lds) && \
    __has_builtin(__builtin_amdgcn_s_wait_tensorcnt)
#define HAVE_TDM 1
#else
#define HAVE_TDM 0
#endif

// Weights pre-converted per launch: f16, transposed [Cout][Kpad], zero-padded.
struct LayerH { const _Float16* Wt; const float* b; const float* s; const float* t; };

// ---------------------------------------------------------------------------
// WMMA helpers.
// A fragment (16x32 f16, row-major in LDS): lane L holds row (L&15),
// K-chunks at kb=(L>>4)*8: v[0..7]=K[kb..kb+7], v[8..15]=K[kb+16..kb+23].
// B fragment: mirrored per-lane layout from a [N][K] (Bt) tile.
// C/D 16x16 f32: vgpr r, lane l -> m = r + 8*(l>>4), n = (l&15).
// ---------------------------------------------------------------------------
__device__ inline v16h load_frag(const _Float16* p, int stride, int lane) {
  int row = lane & 15;
  int kb  = (lane >> 4) * 8;
  const _Float16* r0 = p + row * stride + kb;
  v16h v;
#pragma unroll
  for (int i = 0; i < 8; i++) { v[i] = r0[i]; v[i + 8] = r0[16 + i]; }
  return v;
}

__device__ inline v8f wmma32(v16h a, v16h b, v8f c) {
  return __builtin_amdgcn_wmma_f32_16x16x32_f16(false, a, false, b, (short)0, c,
                                                false, false);
}

// ---------------------------------------------------------------------------
// TDM issue of a [COUT][32] f16 Bt tile into LDS from the packed [COUT][Kpad]
// weight tensor at column k0 (2D D#: data_size=2B, tile_dim0=32,
// tile_dim1=COUT, dim0_stride=Kpad). No wait, no barrier: caller sequences.
// ---------------------------------------------------------------------------
#if HAVE_TDM
__device__ inline void tdm_issue_w_tile(_Float16* dst,
                                        const _Float16* __restrict__ WtG,
                                        int Kpad, int COUT, int k0) {
  unsigned lds = (unsigned)(size_t)(void*)dst;
  unsigned long long ga =
      (unsigned long long)(size_t)(const void*)WtG + (unsigned long long)k0 * 2ull;
  v4u g0;
  g0[0] = 1u;                                   // count=1, user descriptor
  g0[1] = lds;                                  // lds_addr
  g0[2] = (unsigned)ga;                         // global_addr[31:0]
  g0[3] = ((unsigned)(ga >> 32) & 0x01FFFFFFu)  // global_addr[56:32]
          | (2u << 30);                         // type=2 (image)
  v8i g1;
  g1[0] = (int)(1u << 16);                                         // data_size=2B
  g1[1] = (int)(((unsigned)Kpad & 0xFFFFu) << 16);                 // tensor_dim0 lo
  g1[2] = (int)((((unsigned)Kpad >> 16) & 0xFFFFu) |
                (((unsigned)COUT & 0xFFFFu) << 16));               // dim0 hi | dim1 lo
  g1[3] = (int)((((unsigned)COUT >> 16) & 0xFFFFu) | (32u << 16)); // dim1 hi | tile_dim0
  g1[4] = (int)((unsigned)COUT & 0xFFFFu);                         // tile_dim1 | tile_dim2=0
  g1[5] = (int)(unsigned)Kpad;                                     // dim0_stride lo
  g1[6] = 0;
  g1[7] = 0;
  v4i gz; gz[0] = 0; gz[1] = 0; gz[2] = 0; gz[3] = 0;
#if defined(__clang_major__) && (__clang_major__ >= 23)
  v8i gz8; for (int i = 0; i < 8; i++) gz8[i] = 0;
  __builtin_amdgcn_tensor_load_to_lds(g0, g1, gz, gz, gz8, 0);
#else
  __builtin_amdgcn_tensor_load_to_lds(g0, g1, gz, gz, 0);
#endif
}
#endif

// Manual fallback: all threads copy one tile in 16B chunks (no sync inside).
__device__ inline void copy_w_tile(_Float16* dst, const _Float16* __restrict__ WtG,
                                   int Kpad, int COUT, int k0, int tid) {
  int chunks = COUT * 4;
  for (int e = tid; e < chunks; e += 256) {
    int n = e >> 2, p = e & 3;
    *(uint4*)(dst + n * 32 + p * 8) =
        *(const uint4*)(WtG + (size_t)n * Kpad + k0 + p * 8);
  }
}

// ---------------------------------------------------------------------------
// One MLP layer inside the fused SA kernel. Xin: [ROWS][KP] f16 LDS.
// Yout: [ROWS][COUT] f16 LDS. Wt: double-buffered [2][COUT*32] LDS.
// TDM path double-buffers weight tiles: issue k+1, wait tensorcnt<=1 (in-order
// completion guarantees tile k landed), compute overlaps the next fetch.
// ---------------------------------------------------------------------------
template <int ROWS, int KP, int COUT>
__device__ inline void mlp_layer(const _Float16* Xin, _Float16* Yout,
                                 _Float16* Wt, LayerH L, int tid) {
  constexpr int MT = ROWS / 16, NT = COUT / 16, TILES = MT * NT;
  constexpr int PERW = (TILES + 7) / 8;
  constexpr int NKT = KP / 32;
  int wave = tid >> 5, lane = tid & 31;
  v8f acc[PERW];
#pragma unroll
  for (int i = 0; i < PERW; i++)
#pragma unroll
    for (int r = 0; r < 8; r++) acc[i][r] = 0.f;

#if HAVE_TDM
  if (wave == 0) tdm_issue_w_tile(Wt, L.Wt, KP, COUT, 0);
#endif
  for (int kt = 0; kt < NKT; kt++) {
    int k0 = kt * 32;
    _Float16* cur = Wt + (kt & 1) * (COUT * 32);
#if HAVE_TDM
    if (wave == 0) {
      if (kt + 1 < NKT) {
        tdm_issue_w_tile(Wt + ((kt + 1) & 1) * (COUT * 32), L.Wt, KP, COUT,
                         k0 + 32);
        __builtin_amdgcn_s_wait_tensorcnt((short)1);
      } else {
        __builtin_amdgcn_s_wait_tensorcnt((short)0);
      }
    }
    __syncthreads();
#else
    copy_w_tile(cur, L.Wt, KP, COUT, k0, tid);
    __syncthreads();
#endif
#pragma unroll
    for (int i = 0; i < PERW; i++) {
      int t = wave + i * 8;
      if (t < TILES) {
        int mt = t / NT, nt = t % NT;
        v16h a = load_frag(Xin + (mt * 16) * KP + k0, KP, lane);
        v16h b = load_frag(cur + (nt * 16) * 32, 32, lane);
        acc[i] = wmma32(a, b, acc[i]);
      }
    }
    __syncthreads();
  }
#pragma unroll
  for (int i = 0; i < PERW; i++) {
    int t = wave + i * 8;
    if (t < TILES) {
      int mt = t / NT, nt = t % NT;
      int n = nt * 16 + (lane & 15);
      float Bv = L.b[n], Sv = L.s[n], Tv = L.t[n];
#pragma unroll
      for (int r = 0; r < 8; r++) {
        int m = mt * 16 + r + 8 * (lane >> 4);
        float v = (acc[i][r] + Bv) * Sv + Tv;
        Yout[m * COUT + n] = (_Float16)(v > 0.f ? v : 0.f);
      }
    }
  }
  __syncthreads();
}

// ---------------------------------------------------------------------------
// Fused SA-MSG kernel: one block per (batch, query point). Gathers k neighbors
// (features + relative xyz) into LDS f16 (vectorized 16B chunks), runs the
// 3-layer WMMA MLP, max-pools over neighbors. PASSES>1 splits k rows.
// ---------------------------------------------------------------------------
template <int ROWS, int PASSES, int CF, int CIN, int C0, int C1, int C2>
__global__ void __launch_bounds__(256) sa_fused_kernel(
    const int* __restrict__ gidx, const float* __restrict__ pts,
    const float* __restrict__ xyz, const float* __restrict__ nxyz,
    int Npts, int NP, LayerH L0, LayerH L1, LayerH L2,
    float* __restrict__ out, int outStride, int outOff) {
  constexpr int P0 = ((CIN + 31) / 32) * 32;
  constexpr int PA = (P0 > C1) ? P0 : C1;
  constexpr int PB = (C0 > C2) ? C0 : C2;
  constexpr int CW = (C0 > C1) ? ((C0 > C2) ? C0 : C2) : ((C1 > C2) ? C1 : C2);
  __shared__ __align__(16) _Float16 bufA[ROWS * PA];
  __shared__ __align__(16) _Float16 bufB[ROWS * PB];
  __shared__ __align__(16) _Float16 bufW[2 * CW * 32];

  int b = blockIdx.x / NP, q = blockIdx.x % NP;
  int tid = threadIdx.x;
  const int* gi = gidx + ((size_t)b * NP + q) * (ROWS * PASSES);
  float cmax = -1e30f;

  for (int pass = 0; pass < PASSES; pass++) {
    constexpr int CHUNKS = P0 / 8;  // 8-half (16B) chunks per row
    for (int e = tid; e < ROWS * CHUNKS; e += 256) {
      int j = e / CHUNKS, c0 = (e - j * CHUNKS) * 8;
      int g = gi[pass * ROWS + j];
      const float* src = pts + ((size_t)b * Npts + g) * CF;
      __align__(16) _Float16 tmp[8];
      if (c0 + 7 < CF) {  // fully inside feature channels: 16B-aligned loads
        const float4* s4 = (const float4*)(src + c0);
        float4 a = s4[0], bb = s4[1];
        tmp[0] = (_Float16)a.x;  tmp[1] = (_Float16)a.y;
        tmp[2] = (_Float16)a.z;  tmp[3] = (_Float16)a.w;
        tmp[4] = (_Float16)bb.x; tmp[5] = (_Float16)bb.y;
        tmp[6] = (_Float16)bb.z; tmp[7] = (_Float16)bb.w;
      } else {
#pragma unroll
        for (int u = 0; u < 8; u++) {
          int c = c0 + u;
          float v = 0.f;
          if (c < CF) v = src[c];
          else if (c < CIN) {
            int d = c - CF;
            v = xyz[((size_t)b * Npts + g) * 3 + d] -
                nxyz[((size_t)b * NP + q) * 3 + d];
          }
          tmp[u] = (_Float16)v;
        }
      }
      *(uint4*)(&bufA[j * P0 + c0]) = *(const uint4*)tmp;
    }
    __syncthreads();
    mlp_layer<ROWS, P0, C0>(bufA, bufB, bufW, L0, tid);
    mlp_layer<ROWS, C0, C1>(bufB, bufA, bufW, L1, tid);
    mlp_layer<ROWS, C1, C2>(bufA, bufB, bufW, L2, tid);
    if (tid < C2) {
      float m = cmax;
      for (int j = 0; j < ROWS; j++) {
        float v = (float)bufB[j * C2 + tid];
        m = v > m ? v : m;
      }
      cmax = m;
    }
    __syncthreads();
  }
  if (tid < C2) out[((size_t)b * NP + q) * outStride + outOff + tid] = cmax;
}

// ---------------------------------------------------------------------------
// Generic WMMA GEMM: Y[M,N] = relu((X[M,K] @ W + b) * s + t), W packed f16
// [N][Kpad]. 64x64 tile, 256 threads (8 waves x 2 wmma tiles), K tiled by 32,
// double-buffered weight tiles via TDM. N is always a multiple of 64 here.
// ---------------------------------------------------------------------------
__global__ void __launch_bounds__(256) gemm_affine_kernel(
    const float* __restrict__ X, const _Float16* __restrict__ Wt,
    const float* __restrict__ bb, const float* __restrict__ ss,
    const float* __restrict__ tt, float* __restrict__ Y,
    int M, int N, int K, int Kpad) {
  __shared__ __align__(16) _Float16 Xs[64 * 32];
  __shared__ __align__(16) _Float16 Ws[2 * 64 * 32];
  int tid = threadIdx.x, wave = tid >> 5, lane = tid & 31;
  int m0 = blockIdx.x * 64, n0 = blockIdx.y * 64;
  v8f acc0, acc1;
#pragma unroll
  for (int r = 0; r < 8; r++) { acc0[r] = 0.f; acc1[r] = 0.f; }
  int t0 = wave * 2, t1 = wave * 2 + 1;
  int mt0 = t0 >> 2, nt0 = t0 & 3, mt1 = t1 >> 2, nt1 = t1 & 3;
  const _Float16* WtBase = Wt + (size_t)n0 * Kpad;
  int NKT = Kpad >> 5;

#if HAVE_TDM
  if (wave == 0) tdm_issue_w_tile(Ws, WtBase, Kpad, 64, 0);
#endif
  for (int kt = 0; kt < NKT; kt++) {
    int k0 = kt * 32;
    // stage X tile (f32 -> f16), one 8-wide row segment per thread; the
    // in-bounds fast path issues 8 unguarded coalesced loads (single wait).
    {
      int r = tid >> 2, part = tid & 3;
      int mg = m0 + r, ck = k0 + part * 8;
      __align__(16) _Float16 tmp[8];
      if (mg < M && ck + 7 < K) {
        const float* src = X + (size_t)mg * K + ck;
#pragma unroll
        for (int u = 0; u < 8; u++) tmp[u] = (_Float16)src[u];
      } else {
#pragma unroll
        for (int u = 0; u < 8; u++) {
          int kg = ck + u;
          tmp[u] = (mg < M && kg < K) ? (_Float16)X[(size_t)mg * K + kg]
                                      : (_Float16)0.f;
        }
      }
      *(uint4*)(&Xs[r * 32 + part * 8]) = *(const uint4*)tmp;
      if (mg < M) __builtin_prefetch(X + (size_t)mg * K + k0 + 32, 0, 1);
    }
    _Float16* cur = Ws + (kt & 1) * (64 * 32);
#if HAVE_TDM
    if (wave == 0) {
      if (kt + 1 < NKT) {
        tdm_issue_w_tile(Ws + ((kt + 1) & 1) * (64 * 32), WtBase, Kpad, 64,
                         k0 + 32);
        __builtin_amdgcn_s_wait_tensorcnt((short)1);
      } else {
        __builtin_amdgcn_s_wait_tensorcnt((short)0);
      }
    }
    __syncthreads();
#else
    copy_w_tile(cur, WtBase, Kpad, 64, k0, tid);
    __syncthreads();
#endif
    v16h a0 = load_frag(Xs + mt0 * 16 * 32, 32, lane);
    v16h b0 = load_frag(cur + nt0 * 16 * 32, 32, lane);
    acc0 = wmma32(a0, b0, acc0);
    v16h a1 = (mt1 == mt0) ? a0 : load_frag(Xs + mt1 * 16 * 32, 32, lane);
    v16h b1 = load_frag(cur + nt1 * 16 * 32, 32, lane);
    acc1 = wmma32(a1, b1, acc1);
    __syncthreads();
  }
  auto emit = [&](v8f acc, int mt, int nt) {
    int n = n0 + nt * 16 + (lane & 15);
    if (n >= N) return;
    float Bv = bb[n], Sv = ss[n], Tv = tt[n];
#pragma unroll
    for (int r = 0; r < 8; r++) {
      int m = m0 + mt * 16 + r + 8 * (lane >> 4);
      if (m < M) {
        float v = (acc[r] + Bv) * Sv + Tv;
        Y[(size_t)m * N + n] = v > 0.f ? v : 0.f;
      }
    }
  };
  emit(acc0, mt0, nt0);
  emit(acc1, mt1, nt1);
}

// ---------------------------------------------------------------------------
// Weight repack: W[K][N] f32 -> Wt[N][Kpad] f16, zero-padded K tail.
// ---------------------------------------------------------------------------
__global__ void convert_wt_kernel(const float* __restrict__ W,
                                  _Float16* __restrict__ Wt, int K, int N,
                                  int Kpad) {
  int i = blockIdx.x * blockDim.x + threadIdx.x;
  if (i >= N * Kpad) return;
  int n = i / Kpad, k = i - n * Kpad;
  Wt[i] = (k < K) ? (_Float16)W[(size_t)k * N + n] : (_Float16)0.f;
}

// ---------------------------------------------------------------------------
// Farthest point sampling: one block per batch, serial scan with block argmax.
// Emits carry-before-update (matches lax.scan output), far starts at 0.
// ---------------------------------------------------------------------------
template <int NPTS>
__global__ void __launch_bounds__(1024) fps_kernel(const float* __restrict__ xyz,
                                                   int* __restrict__ outIdx,
                                                   int npoint) {
  __shared__ float dist[NPTS];
  __shared__ float rv[1024];
  __shared__ int ri[1024];
  __shared__ float cent[3];
  int b = blockIdx.x, tid = threadIdx.x, T = blockDim.x;
  const float* X = xyz + (size_t)b * NPTS * 3;
  for (int i = tid; i < NPTS; i += T) dist[i] = 1e10f;
  int far = 0;
  __syncthreads();
  for (int s = 0; s < npoint; s++) {
    if (tid == 0) outIdx[b * npoint + s] = far;
    if (s == npoint - 1) break;
    if (tid < 3) cent[tid] = X[far * 3 + tid];
    __syncthreads();
    float bv = -1.f; int bi = 0;
    for (int i = tid; i < NPTS; i += T) {
      float dx = X[i * 3 + 0] - cent[0];
      float dy = X[i * 3 + 1] - cent[1];
      float dz = X[i * 3 + 2] - cent[2];
      float nd = fminf(dist[i], dx * dx + dy * dy + dz * dz);
      dist[i] = nd;
      if (nd > bv) { bv = nd; bi = i; }
    }
    rv[tid] = bv; ri[tid] = bi;
    __syncthreads();
    for (int off = T / 2; off > 0; off >>= 1) {
      if (tid < off) {
        if (rv[tid + off] > rv[tid] ||
            (rv[tid + off] == rv[tid] && ri[tid + off] < ri[tid])) {
          rv[tid] = rv[tid + off]; ri[tid] = ri[tid + off];
        }
      }
      __syncthreads();
    }
    far = ri[0];
    __syncthreads();
  }
}

// ---------------------------------------------------------------------------
// Ball query: one wave32 per query point; ordered stream compaction of the
// first K in-radius indices via ballot masks; pad tail with first hit.
// ---------------------------------------------------------------------------
template <int K>
__global__ void __launch_bounds__(256) query_ball_kernel(
    const float* __restrict__ xyz, const float* __restrict__ nxyz,
    int* __restrict__ gidx, int Npts, int NP, int Btot, float r2) {
  int gw = blockIdx.x * (blockDim.x >> 5) + (threadIdx.x >> 5);
  int lane = threadIdx.x & 31;
  if (gw >= Btot * NP) return;
  int b = gw / NP;
  float qx = nxyz[(size_t)gw * 3 + 0];
  float qy = nxyz[(size_t)gw * 3 + 1];
  float qz = nxyz[(size_t)gw * 3 + 2];
  int* out = gidx + (size_t)gw * K;
  int cnt = 0, first = 0;
  for (int base = 0; base < Npts && cnt < K; base += 32) {
    int i = base + lane;
    bool pred = false;
    if (i < Npts) {
      float dx = xyz[((size_t)b * Npts + i) * 3 + 0] - qx;
      float dy = xyz[((size_t)b * Npts + i) * 3 + 1] - qy;
      float dz = xyz[((size_t)b * Npts + i) * 3 + 2] - qz;
      pred = (dx * dx + dy * dy + dz * dz) <= r2;
    }
    unsigned mask = (unsigned)__ballot(pred);
    if (cnt == 0 && mask) {
      int fl = __ffs((int)mask) - 1;
      first = __shfl(i, fl, 32);
    }
    if (pred) {
      int pos = cnt + __popc(mask & ((1u << lane) - 1u));
      if (pos < K) out[pos] = i;
    }
    cnt += __popc(mask);
  }
  if (cnt < K)
    for (int j = cnt + lane; j < K; j += 32) out[j] = first;
}

// --------------------------- small utility kernels -------------------------
__global__ void transpose_split_kernel(const float* __restrict__ x,
                                       float* __restrict__ xyz,
                                       float* __restrict__ feat,
                                       float* __restrict__ pts, int Bc, int Nc) {
  int i = blockIdx.x * blockDim.x + threadIdx.x;
  if (i >= Bc * Nc) return;
  int b = i / Nc, n = i - b * Nc;
  for (int c = 0; c < 3; c++) {
    float vx = x[((size_t)b * 6 + c) * Nc + n];
    float vf = x[((size_t)b * 6 + 3 + c) * Nc + n];
    xyz[(size_t)i * 3 + c] = vx;
    feat[(size_t)i * 3 + c] = vf;
    pts[(size_t)i * 6 + c] = vx;
    pts[(size_t)i * 6 + 3 + c] = vf;
  }
}

__global__ void gather_xyz_kernel(const float* __restrict__ xyz,
                                  const int* __restrict__ idx,
                                  float* __restrict__ out, int Npts, int NP,
                                  int Btot) {
  int i = blockIdx.x * blockDim.x + threadIdx.x;
  if (i >= Btot * NP) return;
  int b = i / NP;
  int g = idx[i];
  for (int c = 0; c < 3; c++)
    out[(size_t)i * 3 + c] = xyz[((size_t)b * Npts + g) * 3 + c];
}

__global__ void concat2_kernel(const float* __restrict__ A, int Ca,
                               const float* __restrict__ Bp, int Cb,
                               float* __restrict__ out, int rows) {
  int C = Ca + Cb;
  int i = blockIdx.x * blockDim.x + threadIdx.x;
  if (i >= rows * C) return;
  int r = i / C, c = i - r * C;
  out[i] = (c < Ca) ? A[(size_t)r * Ca + c] : Bp[(size_t)r * Cb + (c - Ca)];
}

__global__ void concat_bcast_kernel(const float* __restrict__ A, int Ca,
                                    const float* __restrict__ g, int Cg,
                                    float* __restrict__ out, int Bc, int S) {
  int C = Ca + Cg;
  int i = blockIdx.x * blockDim.x + threadIdx.x;
  if (i >= Bc * S * C) return;
  int r = i / C, c = i - r * C;
  int b = r / S;
  out[i] = (c < Ca) ? A[(size_t)r * Ca + c] : g[(size_t)b * Cg + (c - Ca)];
}

__global__ void rowmax_kernel(const float* __restrict__ X, float* __restrict__ out,
                              int Bc, int S, int C) {
  int b = blockIdx.x;
  for (int c = threadIdx.x; c < C; c += blockDim.x) {
    float m = -1e30f;
    for (int s = 0; s < S; s++) m = fmaxf(m, X[((size_t)b * S + s) * C + c]);
    out[(size_t)b * C + c] = m;
  }
}

__global__ void knn3_kernel(const float* __restrict__ xyz1,
                            const float* __restrict__ xyz2,
                            int* __restrict__ idx3, float* __restrict__ w3,
                            int Bc, int N1, int S) {
  int i = blockIdx.x * blockDim.x + threadIdx.x;
  if (i >= Bc * N1) return;
  int b = i / N1;
  float px = xyz1[(size_t)i * 3], py = xyz1[(size_t)i * 3 + 1],
        pz = xyz1[(size_t)i * 3 + 2];
  const float* Q = xyz2 + (size_t)b * S * 3;
  float d0 = 1e30f, d1 = 1e30f, d2 = 1e30f;
  int i0 = 0, i1 = 0, i2 = 0;
  for (int s = 0; s < S; s++) {
    float dx = Q[s * 3] - px, dy = Q[s * 3 + 1] - py, dz = Q[s * 3 + 2] - pz;
    float d = dx * dx + dy * dy + dz * dz;
    if (d < d0)      { d2 = d1; i2 = i1; d1 = d0; i1 = i0; d0 = d; i0 = s; }
    else if (d < d1) { d2 = d1; i2 = i1; d1 = d; i1 = s; }
    else if (d < d2) { d2 = d; i2 = s; }
  }
  float r0 = 1.f / (d0 + 1e-8f), r1 = 1.f / (d1 + 1e-8f), r2 = 1.f / (d2 + 1e-8f);
  float sum = r0 + r1 + r2;
  idx3[i * 3] = i0; idx3[i * 3 + 1] = i1; idx3[i * 3 + 2] = i2;
  w3[i * 3] = r0 / sum; w3[i * 3 + 1] = r1 / sum; w3[i * 3 + 2] = r2 / sum;
}

__global__ void interp_cat_kernel(const float* __restrict__ P1, int C1,
                                  const float* __restrict__ P2, int C2,
                                  const int* __restrict__ idx3,
                                  const float* __restrict__ w3,
                                  float* __restrict__ out, int Bc, int N1, int S) {
  int C = C1 + C2;
  int i = blockIdx.x * blockDim.x + threadIdx.x;
  if (i >= Bc * N1 * C) return;
  int r = i / C, c = i - r * C;
  if (c < C1) { out[i] = P1[(size_t)r * C1 + c]; return; }
  int b = r / N1, cc = c - C1;
  const int* id = idx3 + r * 3;
  const float* wv = w3 + r * 3;
  float acc = 0.f;
  for (int j = 0; j < 3; j++)
    acc += wv[j] * P2[((size_t)b * S + id[j]) * C2 + cc];
  out[i] = acc;
}

__global__ void head_final_kernel(const float* __restrict__ H,
                                  const float* __restrict__ W2,
                                  const float* __restrict__ b2,
                                  float* __restrict__ out, int rows) {
  int i = blockIdx.x * blockDim.x + threadIdx.x;
  if (i >= rows) return;
  float acc = b2[0];
  for (int c = 0; c < 128; c++) acc += H[(size_t)i * 128 + c] * W2[c];
  out[i] = 1.f / (1.f + expf(-acc));
}

// ---------------------------------------------------------------------------
static inline size_t alignUp(size_t x) { return (x + 255) & ~(size_t)255; }
static inline int ceilDiv(int a, int b) { return (a + b - 1) / b; }

extern "C" void kernel_launch(void* const* d_in, const int* in_sizes, int n_in,
                              void* d_out, int out_size, void* d_ws,
                              size_t ws_size, hipStream_t stream) {
  (void)in_sizes; (void)n_in; (void)out_size; (void)ws_size;
  const float* x = (const float*)d_in[0];
  auto F = [&](int i) { return (const float*)d_in[i]; };
  // pytree (sorted-key) param index map:
  // fp1: L0@2 L1@6 | fp2: L0@10 L1@14 | fp3: L0@18 L1@22
  // head1: W2@26 b2@27 c1@28 | head2: W2@32 b2@33 c1@34
  // sa1: s0{38,42,46} s1{50,54,58} s2{62,66,70}
  // sa2: s0{74,78,82} s1{86,90,94} s2{98,102,106}
  // sa3: L0@110 L1@114 L2@118

  char* wsb = (char*)d_ws;
  size_t off = 0;
  auto allocF = [&](size_t n) { float* p = (float*)(wsb + off); off = alignUp(off + n * 4); return p; };
  auto allocI = [&](size_t n) { int* p = (int*)(wsb + off); off = alignUp(off + n * 4); return p; };
  auto allocH = [&](size_t n) { _Float16* p = (_Float16*)(wsb + off); off = alignUp(off + n * 2); return p; };

  float* xyz0  = allocF((size_t)BATCH * NPTS0 * 3);
  float* feat0 = allocF((size_t)BATCH * NPTS0 * 3);
  float* pts0  = allocF((size_t)BATCH * NPTS0 * 6);
  int*   fps1  = allocI((size_t)BATCH * NP1);
  float* nx1   = allocF((size_t)BATCH * NP1 * 3);
  int*   gidx  = allocI((size_t)BATCH * NP1 * 128);
  float* l1pts = allocF((size_t)BATCH * NP1 * 320);
  int*   fps2  = allocI((size_t)BATCH * NP2);
  float* nx2   = allocF((size_t)BATCH * NP2 * 3);
  float* l2pts = allocF((size_t)BATCH * NP2 * 640);
  float* catb  = allocF((size_t)BATCH * 512 * 1664);
  float* tmpA  = allocF((size_t)2048 * 1024 + 256);
  float* tmpB  = allocF((size_t)2048 * 1024 + 256);
  float* l3    = allocF((size_t)BATCH * 1024);
  float* l2fp  = allocF((size_t)BATCH * 512 * 256);
  int*   knnI  = allocI((size_t)BATCH * NPTS0 * 3);
  float* knnW  = allocF((size_t)BATCH * NPTS0 * 3);
  float* l1fp  = allocF((size_t)BATCH * NP1 * 128);
  float* l0    = allocF((size_t)BATCH * NPTS0 * 128);
  float* hbuf  = allocF((size_t)BATCH * NPTS0 * 128);

  // Repack every layer's weights once per launch: f16 [N][Kpad], zero-padded.
  struct LayerHK { LayerH L; int Kpad; };
  auto mkH = [&](int base, int K, int N) {
    int Kpad = (K + 31) & ~31;
    _Float16* wt = allocH((size_t)N * Kpad);
    convert_wt_kernel<<<ceilDiv(N * Kpad, 256), 256, 0, stream>>>(
        F(base), wt, K, N, Kpad);
    LayerHK r; r.L = LayerH{wt, F(base + 1), F(base + 2), F(base + 3)};
    r.Kpad = Kpad;
    return r;
  };

  LayerHK sa1_0a = mkH(38, 6, 32),   sa1_0b = mkH(42, 32, 32),  sa1_0c = mkH(46, 32, 64);
  LayerHK sa1_1a = mkH(50, 6, 64),   sa1_1b = mkH(54, 64, 64),  sa1_1c = mkH(58, 64, 128);
  LayerHK sa1_2a = mkH(62, 6, 64),   sa1_2b = mkH(66, 64, 96),  sa1_2c = mkH(70, 96, 128);
  LayerHK sa2_0a = mkH(74, 323, 64), sa2_0b = mkH(78, 64, 64),  sa2_0c = mkH(82, 64, 128);
  LayerHK sa2_1a = mkH(86, 323, 128),sa2_1b = mkH(90, 128, 128),sa2_1c = mkH(94, 128, 256);
  LayerHK sa2_2a = mkH(98, 323, 128),sa2_2b = mkH(102,128, 128),sa2_2c = mkH(106,128, 256);
  LayerHK sa3a = mkH(110, 643, 256), sa3b = mkH(114, 256, 512), sa3c = mkH(118, 512, 1024);
  LayerHK fp3a = mkH(18, 1664, 256), fp3b = mkH(22, 256, 256);
  LayerHK fp2a = mkH(10, 576, 256),  fp2b = mkH(14, 256, 128);
  LayerHK fp1a = mkH(2, 134, 128),   fp1b = mkH(6, 128, 128);
  LayerHK h1c = mkH(28, 128, 128),   h2c = mkH(34, 128, 128);

  auto gemm = [&](const float* X, LayerHK L, float* Y, int M, int N, int K) {
    dim3 g(ceilDiv(M, 64), ceilDiv(N, 64));
    gemm_affine_kernel<<<g, 256, 0, stream>>>(X, L.L.Wt, L.L.b, L.L.s, L.L.t,
                                              Y, M, N, K, L.Kpad);
  };

  // 0) split input
  transpose_split_kernel<<<ceilDiv(BATCH * NPTS0, 256), 256, 0, stream>>>(
      x, xyz0, feat0, pts0, BATCH, NPTS0);

  // 1) SA1
  fps_kernel<NPTS0><<<BATCH, 1024, 0, stream>>>(xyz0, fps1, NP1);
  gather_xyz_kernel<<<ceilDiv(BATCH * NP1, 256), 256, 0, stream>>>(
      xyz0, fps1, nx1, NPTS0, NP1, BATCH);
  int qb1 = ceilDiv(BATCH * NP1, 8);
  query_ball_kernel<32><<<qb1, 256, 0, stream>>>(xyz0, nx1, gidx, NPTS0, NP1, BATCH, 0.001f * 0.001f);
  sa_fused_kernel<32, 1, 3, 6, 32, 32, 64><<<BATCH * NP1, 256, 0, stream>>>(
      gidx, feat0, xyz0, nx1, NPTS0, NP1, sa1_0a.L, sa1_0b.L, sa1_0c.L, l1pts, 320, 0);
  query_ball_kernel<64><<<qb1, 256, 0, stream>>>(xyz0, nx1, gidx, NPTS0, NP1, BATCH, 0.002f * 0.002f);
  sa_fused_kernel<64, 1, 3, 6, 64, 64, 128><<<BATCH * NP1, 256, 0, stream>>>(
      gidx, feat0, xyz0, nx1, NPTS0, NP1, sa1_1a.L, sa1_1b.L, sa1_1c.L, l1pts, 320, 64);
  query_ball_kernel<128><<<qb1, 256, 0, stream>>>(xyz0, nx1, gidx, NPTS0, NP1, BATCH, 0.004f * 0.004f);
  sa_fused_kernel<64, 2, 3, 6, 64, 96, 128><<<BATCH * NP1, 256, 0, stream>>>(
      gidx, feat0, xyz0, nx1, NPTS0, NP1, sa1_2a.L, sa1_2b.L, sa1_2c.L, l1pts, 320, 192);

  // 2) SA2
  fps_kernel<NP1><<<BATCH, 1024, 0, stream>>>(nx1, fps2, NP2);
  gather_xyz_kernel<<<ceilDiv(BATCH * NP2, 256), 256, 0, stream>>>(
      nx1, fps2, nx2, NP1, NP2, BATCH);
  int qb2 = ceilDiv(BATCH * NP2, 8);
  query_ball_kernel<64><<<qb2, 256, 0, stream>>>(nx1, nx2, gidx, NP1, NP2, BATCH, 0.004f * 0.004f);
  sa_fused_kernel<64, 1, 320, 323, 64, 64, 128><<<BATCH * NP2, 256, 0, stream>>>(
      gidx, l1pts, nx1, nx2, NP1, NP2, sa2_0a.L, sa2_0b.L, sa2_0c.L, l2pts, 640, 0);
  query_ball_kernel<64><<<qb2, 256, 0, stream>>>(nx1, nx2, gidx, NP1, NP2, BATCH, 0.006f * 0.006f);
  sa_fused_kernel<64, 1, 320, 323, 128, 128, 256><<<BATCH * NP2, 256, 0, stream>>>(
      gidx, l1pts, nx1, nx2, NP1, NP2, sa2_1a.L, sa2_1b.L, sa2_1c.L, l2pts, 640, 128);
  query_ball_kernel<128><<<qb2, 256, 0, stream>>>(nx1, nx2, gidx, NP1, NP2, BATCH, 0.008f * 0.008f);
  sa_fused_kernel<64, 2, 320, 323, 128, 128, 256><<<BATCH * NP2, 256, 0, stream>>>(
      gidx, l1pts, nx1, nx2, NP1, NP2, sa2_2a.L, sa2_2b.L, sa2_2c.L, l2pts, 640, 384);

  // 3) SA3 (global)
  concat2_kernel<<<ceilDiv(BATCH * 512 * 643, 256), 256, 0, stream>>>(
      nx2, 3, l2pts, 640, catb, BATCH * 512);
  gemm(catb, sa3a, tmpA, BATCH * 512, 256, 643);
  gemm(tmpA, sa3b, tmpB, BATCH * 512, 512, 256);
  gemm(tmpB, sa3c, tmpA, BATCH * 512, 1024, 512);
  rowmax_kernel<<<BATCH, 256, 0, stream>>>(tmpA, l3, BATCH, 512, 1024);

  // 4) FP3 (S==1 broadcast)
  concat_bcast_kernel<<<ceilDiv(BATCH * 512 * 1664, 256), 256, 0, stream>>>(
      l2pts, 640, l3, 1024, catb, BATCH, 512);
  gemm(catb, fp3a, tmpB, BATCH * 512, 256, 1664);
  gemm(tmpB, fp3b, l2fp, BATCH * 512, 256, 256);

  // 5) FP2 (3-NN interp 512 -> 1000)
  knn3_kernel<<<ceilDiv(BATCH * NP1, 256), 256, 0, stream>>>(nx1, nx2, knnI, knnW, BATCH, NP1, NP2);
  interp_cat_kernel<<<ceilDiv(BATCH * NP1 * 576, 256), 256, 0, stream>>>(
      l1pts, 320, l2fp, 256, knnI, knnW, catb, BATCH, NP1, NP2);
  gemm(catb, fp2a, tmpA, BATCH * NP1, 256, 576);
  gemm(tmpA, fp2b, l1fp, BATCH * NP1, 128, 256);

  // 6) FP1 (3-NN interp 1000 -> 3000)
  knn3_kernel<<<ceilDiv(BATCH * NPTS0, 256), 256, 0, stream>>>(xyz0, nx1, knnI, knnW, BATCH, NPTS0, NP1);
  interp_cat_kernel<<<ceilDiv(BATCH * NPTS0 * 134, 256), 256, 0, stream>>>(
      pts0, 6, l1fp, 128, knnI, knnW, catb, BATCH, NPTS0, NP1);
  gemm(catb, fp1a, tmpA, BATCH * NPTS0, 128, 134);
  gemm(tmpA, fp1b, l0, BATCH * NPTS0, 128, 128);

  // 7) heads
  float* out = (float*)d_out;
  gemm(l0, h1c, hbuf, BATCH * NPTS0, 128, 128);
  head_final_kernel<<<ceilDiv(BATCH * NPTS0, 256), 256, 0, stream>>>(
      hbuf, F(26), F(27), out, BATCH * NPTS0);
  gemm(l0, h2c, hbuf, BATCH * NPTS0, 128, 128);
  head_final_kernel<<<ceilDiv(BATCH * NPTS0, 256), 256, 0, stream>>>(
      hbuf, F(32), F(33), out + BATCH * NPTS0, BATCH * NPTS0);
}
